// QueryInitializer_78005196030102
// MI455X (gfx1250) — matrix operand order repaired and checked by reading, not compiled, not run on hardware
//
#include <hip/hip_runtime.h>

typedef __attribute__((ext_vector_type(16))) _Float16 v16h;
typedef __attribute__((ext_vector_type(8)))  float    v8f;

#define B_   4
#define N_   131072
#define K_   128
#define F_   96
#define Q_   128
#define FPS_T 1024
#define NPT  (N_ / FPS_T)   // 128 points per thread

// ---------------------------------------------------------------------------
// Kernel 1: furthest point sampling. One block per batch, 1024 threads.
// min_d2 lives in d_ws (L2-resident, 23.3TB/s HBM never touched after warmup).
// Sequential K-1 steps; block argmax with first-index tie-break (jnp.argmax).
// ---------------------------------------------------------------------------
__global__ __launch_bounds__(FPS_T)
void fps_kernel(const float* __restrict__ coords,
                float* __restrict__ md,          // [B_*N_] scratch
                float* __restrict__ out_coords,  // d_out queries tail: [B,K,3]
                float* __restrict__ ws_coords)   // [B,K,3] scratch copy
{
    __shared__ float rv[FPS_T];
    __shared__ int   ri[FPS_T];
    __shared__ float s_last[3];

    const int b   = blockIdx.x;
    const int tid = threadIdx.x;
    const float* cb = coords + (size_t)b * N_ * 3;
    float* mdb = md + (size_t)b * N_;

    if (tid == 0) {
        float x = cb[0], y = cb[1], z = cb[2];
        s_last[0] = x; s_last[1] = y; s_last[2] = z;
        int o = (b * K_ + 0) * 3;
        out_coords[o] = x; out_coords[o + 1] = y; out_coords[o + 2] = z;
        ws_coords[o]  = x; ws_coords[o + 1]  = y; ws_coords[o + 2]  = z;
    }
    __syncthreads();

    for (int k = 1; k < K_; ++k) {
        const float lx = s_last[0], ly = s_last[1], lz = s_last[2];
        float bv = -1.0f; int bi = 0;
        const bool first = (k == 1);
        for (int i = 0; i < NPT; ++i) {
            int p = tid + i * FPS_T;
            float dx = cb[3 * p + 0] - lx;
            float dy = cb[3 * p + 1] - ly;
            float dz = cb[3 * p + 2] - lz;
            float d2 = dx * dx + dy * dy + dz * dz;
            float prev = first ? 1e10f : mdb[p];
            float m = prev < d2 ? prev : d2;
            mdb[p] = m;
            if (m > bv) { bv = m; bi = p; }   // ascending p: keeps first max
        }
        rv[tid] = bv; ri[tid] = bi;
        __syncthreads();
        for (int s = FPS_T / 2; s > 0; s >>= 1) {
            if (tid < s) {
                float ov = rv[tid + s]; int oi = ri[tid + s];
                float mv = rv[tid];     int mi = ri[tid];
                if (ov > mv || (ov == mv && oi < mi)) { rv[tid] = ov; ri[tid] = oi; }
            }
            __syncthreads();
        }
        if (tid == 0) {
            int best = ri[0];
            float x = cb[3 * best], y = cb[3 * best + 1], z = cb[3 * best + 2];
            s_last[0] = x; s_last[1] = y; s_last[2] = z;
            int o = (b * K_ + k) * 3;
            out_coords[o] = x; out_coords[o + 1] = y; out_coords[o + 2] = z;
            ws_coords[o]  = x; ws_coords[o + 1]  = y; ws_coords[o + 2]  = z;
        }
        __syncthreads();
    }
}

// ---------------------------------------------------------------------------
// Kernel 2: feat_idx[b,k] = argmax_n ||coords[b,n]-sampled[b,k]||^2
// 512 independent blocks streaming L2-resident coords.
// ---------------------------------------------------------------------------
__global__ __launch_bounds__(256)
void fargmax_kernel(const float* __restrict__ coords,
                    const float* __restrict__ ws_coords,
                    int* __restrict__ feat_idx)
{
    __shared__ float rv[256];
    __shared__ int   ri[256];
    const int bk  = blockIdx.x;          // b*K_ + k
    const int b   = bk / K_;
    const int tid = threadIdx.x;
    const float* cb = coords + (size_t)b * N_ * 3;

    const float sx = ws_coords[bk * 3 + 0];
    const float sy = ws_coords[bk * 3 + 1];
    const float sz = ws_coords[bk * 3 + 2];

    float bv = -1.0f; int bi = 0;
    for (int p = tid; p < N_; p += 256) {
        float dx = cb[3 * p + 0] - sx;
        float dy = cb[3 * p + 1] - sy;
        float dz = cb[3 * p + 2] - sz;
        float d2 = dx * dx + dy * dy + dz * dz;
        if (d2 > bv) { bv = d2; bi = p; }
    }
    rv[tid] = bv; ri[tid] = bi;
    __syncthreads();
    for (int s = 128; s > 0; s >>= 1) {
        if (tid < s) {
            float ov = rv[tid + s]; int oi = ri[tid + s];
            float mv = rv[tid];     int mi = ri[tid];
            if (ov > mv || (ov == mv && oi < mi)) { rv[tid] = ov; ri[tid] = oi; }
        }
        __syncthreads();
    }
    if (tid == 0) feat_idx[bk] = ri[0];
}

// ---------------------------------------------------------------------------
// Kernel 3: gather features + 2-layer MLP on the WMMA path.
// 32 blocks x 256 threads (8 waves). Each wave owns one 16x16 output tile of
// a 16-row slab. Layer1: K=96 -> 3x v_wmma_f32_16x16x32_f16; Layer2: K=128 ->
// 4x. A-fragments packed from LDS per ISA 7.12.2; B-fragments are strided
// column loads from row-major W (lane = N column, elements = K rows).
// ---------------------------------------------------------------------------
__global__ __launch_bounds__(256)
void mlp_kernel(const float* __restrict__ features,
                const float* __restrict__ W1, const float* __restrict__ b1,
                const float* __restrict__ W2, const float* __restrict__ b2,
                const int* __restrict__ feat_idx,
                float* __restrict__ out_q)
{
    __shared__ _Float16 As[16][F_ + 8];   // f16 activations, layer-1 input
    __shared__ _Float16 Hs[16][Q_ + 8];   // f16 hidden, layer-2 input
    __shared__ int s_idx[16];
    __shared__ int s_b[16];

    const int tid    = threadIdx.x;
    const int wave   = tid >> 5;          // 0..7  -> N-tile
    const int lane   = tid & 31;
    const int laneHi = lane >> 4;         // 0/1
    const int mrow   = lane & 15;
    const int ncol   = wave * 16 + mrow;  // output column this lane owns in C/D
    const int row0   = blockIdx.x * 16;

    if (tid < 16) {
        int r = row0 + tid;
        s_idx[tid] = feat_idx[r];
        s_b[tid]   = r / K_;
    }
    __syncthreads();

    // gather 16 rows x 96 features -> f16 LDS
    for (int e = tid; e < 16 * F_; e += 256) {
        int m = e / F_, f = e - m * F_;
        As[m][f] = (_Float16)features[((size_t)s_b[m] * N_ + s_idx[m]) * F_ + f];
    }
    __syncthreads();

    // ---- layer 1: [16x96] x [96x128] -> 16x128, 3 WMMA K-steps per wave ----
    v8f acc = {0.f, 0.f, 0.f, 0.f, 0.f, 0.f, 0.f, 0.f};
#pragma unroll
    for (int t = 0; t < 3; ++t) {
        v16h a, bf;
#pragma unroll
        for (int j = 0; j < 16; ++j) {
            int v  = j >> 1;
            int kk = 2 * v + ((v >= 4) ? 8 : 0) + laneHi * 8 + (j & 1);
            a[j]  = As[mrow][t * 32 + kk];
            bf[j] = (_Float16)W1[(size_t)(t * 32 + kk) * Q_ + ncol];
        }
        acc = __builtin_amdgcn_wmma_f32_16x16x32_f16(
                  false, a, false, bf, (short)0, acc, false, false);
    }
    const float bias1 = b1[ncol];
#pragma unroll
    for (int g = 0; g < 8; ++g) {
        int m = g + laneHi * 8;
        float h = acc[g] + bias1;
        Hs[m][ncol] = (_Float16)(h > 0.f ? h : 0.f);   // ReLU
    }
    __syncthreads();

    // ---- layer 2: [16x128] x [128x128] -> 16x128, 4 WMMA K-steps ----
    v8f acc2 = {0.f, 0.f, 0.f, 0.f, 0.f, 0.f, 0.f, 0.f};
#pragma unroll
    for (int t = 0; t < 4; ++t) {
        v16h a, bf;
#pragma unroll
        for (int j = 0; j < 16; ++j) {
            int v  = j >> 1;
            int kk = 2 * v + ((v >= 4) ? 8 : 0) + laneHi * 8 + (j & 1);
            a[j]  = Hs[mrow][t * 32 + kk];
            bf[j] = (_Float16)W2[(size_t)(t * 32 + kk) * Q_ + ncol];
        }
        acc2 = __builtin_amdgcn_wmma_f32_16x16x32_f16(
                   false, a, false, bf, (short)0, acc2, false, false);
    }
    const float bias2 = b2[ncol];
#pragma unroll
    for (int g = 0; g < 8; ++g) {
        int m = g + laneHi * 8;
        out_q[(size_t)(row0 + m) * Q_ + ncol] = acc2[g] + bias2;
    }
}

// ---------------------------------------------------------------------------
extern "C" void kernel_launch(void* const* d_in, const int* in_sizes, int n_in,
                              void* d_out, int out_size, void* d_ws, size_t ws_size,
                              hipStream_t stream)
{
    const float* coords   = (const float*)d_in[0];
    const float* features = (const float*)d_in[1];
    const float* W1 = (const float*)d_in[2];
    const float* b1 = (const float*)d_in[3];
    const float* W2 = (const float*)d_in[4];
    const float* b2 = (const float*)d_in[5];

    float* out_q = (float*)d_out;                       // [B,K,Q]
    float* out_c = out_q + (size_t)B_ * K_ * Q_;        // [B,K,3]

    // workspace: min_d2 slab | sampled coords | feat_idx
    float* ws_md     = (float*)d_ws;                                   // B_*N_
    float* ws_coords = ws_md + (size_t)B_ * N_;                        // B_*K_*3
    int*   ws_fidx   = (int*)(ws_coords + (size_t)B_ * K_ * 3);        // B_*K_

    fps_kernel<<<B_, FPS_T, 0, stream>>>(coords, ws_md, out_c, ws_coords);
    fargmax_kernel<<<B_ * K_, 256, 0, stream>>>(coords, ws_coords, ws_fidx);
    mlp_kernel<<<(B_ * K_) / 16, 256, 0, stream>>>(features, W1, b1, W2, b2,
                                                   ws_fidx, out_q);
}
